// Model_75290776699577
// MI455X (gfx1250) — matrix operand order repaired and checked by reading, not compiled, not run on hardware
//
#include <hip/hip_runtime.h>
#include <hip/hip_bf16.h>
#include <stddef.h>

// ---------------- problem constants ----------------
constexpr int Bn = 512;    // batch
constexpr int Tn = 1024;   // time steps
constexpr int In = 80;     // input features
constexpr int Ip = 96;     // input padded to multiple of 32
constexpr int Hn = 160;    // hidden
constexpr int G4 = 4 * Hn; // 640 gate rows
constexpr int NT = Hn / 16;  // 10 waves per block, each owns 16 hidden units
constexpr int KCX = Ip / 32; // 3 K-chunks for layer0 input part
constexpr int KCH = Hn / 32; // 5 K-chunks for hidden parts
constexpr int RS  = 176;     // LDS h row stride (bf16 elems), 352B = 16B-aligned

typedef __bf16 bf16_t;
typedef __attribute__((ext_vector_type(16))) __bf16 v16bf;
typedef __attribute__((ext_vector_type(8)))  __bf16 v8bf;
typedef __attribute__((ext_vector_type(8)))  float  v8f;
typedef __attribute__((ext_vector_type(4)))  float  v4f;

union V16 { v16bf v; v8bf h[2]; };

__device__ __forceinline__ v8f wmma_bf16(v16bf a, v16bf b, v8f c) {
  // D(f32 16x16) = A(16x32 bf16) * B(32x16 bf16) + C
  return __builtin_amdgcn_wmma_f32_16x16x32_bf16(
      /*neg_a=*/false, a, /*neg_b=*/false, b,
      /*c_mod=*/(short)0, c, /*reuse_a=*/false, /*reuse_b=*/false);
}

__device__ __forceinline__ float fsig(float x) {
  return __builtin_amdgcn_rcpf(1.0f + __expf(-x));
}
__device__ __forceinline__ float ftanh_fast(float x) {
  float e = __expf(-2.0f * x);
  return (1.0f - e) * __builtin_amdgcn_rcpf(1.0f + e);
}

// A operand chunk kc from an LDS h row (row = this lane's batch row, bf16):
// runs at kc*32 + hi*8 and kc*32 + hi*8 + 16  (ISA 16-bit A 16x32 layout)
__device__ __forceinline__ v16bf ldA_lds(const bf16_t* rowp, int kc, int hi) {
  V16 u;
  const int K0 = kc * 32 + hi * 8;
  u.h[0] = *(const v8bf*)(rowp + K0);
  u.h[1] = *(const v8bf*)(rowp + K0 + 16);
  return u.v;
}

// 8 contiguous f32 -> v8bf (x staged straight into A registers)
__device__ __forceinline__ v8bf cvt8(const float* __restrict__ p) {
  v4f a = *(const v4f*)p;
  v4f b = *(const v4f*)(p + 4);
  v8bf r;
  r[0] = (bf16_t)a[0]; r[1] = (bf16_t)a[1]; r[2] = (bf16_t)a[2]; r[3] = (bf16_t)a[3];
  r[4] = (bf16_t)b[0]; r[5] = (bf16_t)b[1]; r[6] = (bf16_t)b[2]; r[7] = (bf16_t)b[3];
  return r;
}

// accumulate 4 gate tiles from a pre-swizzled weight stream:
// pw layout = [g][kc][lane][16] bf16  -> per (g,kc): contiguous 1KB per wave.
template <int KC>
__device__ __forceinline__ void accum4(v8f acc[4], const V16* A,
                                       const bf16_t* __restrict__ pw, int lane) {
#pragma unroll
  for (int g = 0; g < 4; ++g) {
#pragma unroll
    for (int kc = 0; kc < KC; ++kc) {
      const bf16_t* p = pw + (((g * KC + kc) << 9) + (lane << 4));
      V16 b;
      b.h[0] = *(const v8bf*)p;
      b.h[1] = *(const v8bf*)(p + 8);
      acc[g] = wmma_bf16(A[kc].v, b.v, acc[g]);
    }
  }
}

// LSTM cell update in the WMMA C/D layout; c stays in registers,
// h_new (bf16) scattered into LDS column j (row m = v + hi*8).
__device__ __forceinline__ void cell_update(const v8f acc[4], v8f& c,
                                            bf16_t* __restrict__ hcol, int hi) {
#pragma unroll
  for (int v = 0; v < 8; ++v) {
    float iv = fsig(acc[0][v]);
    float fv = fsig(acc[1][v]);
    float gv = ftanh_fast(acc[2][v]);
    float ov = fsig(acc[3][v]);
    float cc = fv * c[v] + iv * gv;
    c[v] = cc;
    hcol[(v + hi * 8) * RS] = (bf16_t)(ov * ftanh_fast(cc));
  }
}

// ---------------- weight prep: fp32 -> bf16, swizzled to WMMA-B stream -------
// swizzled index: ((((wv*4 + g)*KC + kc)*32 + lane)*16 + e)
//   maps to source row n = g*H + wv*16 + (lane&15),
//            col       K = kc*32 + (lane&16 ? 8 : 0) + (e&7) + (e&8 ? 16 : 0)
__global__ void lstm_prep(const float* __restrict__ W_ih0, const float* __restrict__ W_hh0,
                          const float* __restrict__ b_ih0, const float* __restrict__ b_hh0,
                          const float* __restrict__ W_ih1, const float* __restrict__ W_hh1,
                          const float* __restrict__ b_ih1, const float* __restrict__ b_hh1,
                          const float* __restrict__ W_out, const float* __restrict__ b_out,
                          bf16_t* __restrict__ W0xs, bf16_t* __restrict__ W0hs,
                          bf16_t* __restrict__ W1xs, bf16_t* __restrict__ W1hs,
                          float* __restrict__ bias0, float* __restrict__ bias1,
                          float* __restrict__ woutf, float* __restrict__ boutf) {
  const int stride = gridDim.x * blockDim.x;
  const int idx0 = blockIdx.x * blockDim.x + threadIdx.x;

  // layer0 x-part weights, K padded 80 -> 96
  for (int i = idx0; i < NT * 4 * KCX * 512; i += stride) {
    int e = i & 15;
    int ln = (i >> 4) & 31;
    int t32 = i >> 9;          // (wv*4+g)*KCX + kc
    int kc = t32 % KCX;
    int gg = t32 / KCX;
    int g = gg & 3, wvv = gg >> 2;
    int n = g * Hn + wvv * 16 + (ln & 15);
    int K = kc * 32 + ((ln & 16) ? 8 : 0) + (e & 7) + ((e & 8) ? 16 : 0);
    W0xs[i] = (K < In) ? (bf16_t)W_ih0[n * In + K] : (bf16_t)0.0f;
  }
  // hidden-sized matrices (W_hh0, W_ih1, W_hh1), K = 160 exact
  for (int i = idx0; i < NT * 4 * KCH * 512; i += stride) {
    int e = i & 15;
    int ln = (i >> 4) & 31;
    int t32 = i >> 9;          // (wv*4+g)*KCH + kc
    int kc = t32 % KCH;
    int gg = t32 / KCH;
    int g = gg & 3, wvv = gg >> 2;
    int n = g * Hn + wvv * 16 + (ln & 15);
    int K = kc * 32 + ((ln & 16) ? 8 : 0) + (e & 7) + ((e & 8) ? 16 : 0);
    W0hs[i] = (bf16_t)W_hh0[n * Hn + K];
    W1xs[i] = (bf16_t)W_ih1[n * Hn + K];
    W1hs[i] = (bf16_t)W_hh1[n * Hn + K];
  }
  for (int i = idx0; i < G4; i += stride) {
    bias0[i] = b_ih0[i] + b_hh0[i];
    bias1[i] = b_ih1[i] + b_hh1[i];
  }
  for (int i = idx0; i < Hn; i += stride) woutf[i] = W_out[i];
  if (idx0 == 0) boutf[0] = b_out[0];
}

// ---------------- persistent 2-layer LSTM kernel -----------------------------
// grid = 32 blocks (one 16-row batch tile each), block = 320 threads (10 waves).
// Wave w owns hidden units [w*16, w*16+16); c-state in VGPRs (WMMA C/D layout),
// h-state double-buffered in LDS (bf16), 2 barriers per timestep. Weights
// stream from L2 every step: the per-iteration opaque byte-offsets (uniform
// SGPRs via readfirstlane) stop LICM from hoisting the loads into spills.
__global__ __launch_bounds__(NT * 32, 1)
void lstm_persistent(const float* __restrict__ x,
                     const bf16_t* __restrict__ W0x, const bf16_t* __restrict__ W0h,
                     const bf16_t* __restrict__ W1x, const bf16_t* __restrict__ W1h,
                     const float* __restrict__ bias0, const float* __restrict__ bias1,
                     const float* __restrict__ wout, const float* __restrict__ bout,
                     float* __restrict__ out) {
  __shared__ bf16_t h0_lds[2][16 * RS];
  __shared__ bf16_t h1_lds[2][16 * RS];

  const int tid  = threadIdx.x;
  const int lane = tid & 31;
  const int wv   = tid >> 5;     // 0..9 : which 16 hidden units this wave owns
  const int nn   = lane & 15;    // column inside tile / batch row for A loads
  const int hi   = lane >> 4;    // half-wave selector in WMMA layouts
  const int b0   = blockIdx.x * 16;

  for (int i = tid; i < 16 * RS; i += blockDim.x) {
    h0_lds[0][i] = (bf16_t)0.0f; h0_lds[1][i] = (bf16_t)0.0f;
    h1_lds[0][i] = (bf16_t)0.0f; h1_lds[1][i] = (bf16_t)0.0f;
  }
  __syncthreads();

  const int j = wv * 16 + nn;  // hidden unit (gate column) this lane owns
  const float bA0[4] = {bias0[0 * Hn + j], bias0[1 * Hn + j],
                        bias0[2 * Hn + j], bias0[3 * Hn + j]};
  const float bA1[4] = {bias1[0 * Hn + j], bias1[1 * Hn + j],
                        bias1[2 * Hn + j], bias1[3 * Hn + j]};

  v8f c0 = {}; v8f c1 = {};

  const float* xrow = x + (size_t)(b0 + nn) * Tn * In;
  const float bout0 = bout[0];

  // per-wave byte offsets into the swizzled weight streams; wv is wave-uniform,
  // readfirstlane makes that explicit so these live in SGPRs.
  const unsigned wvu = (unsigned)__builtin_amdgcn_readfirstlane(wv);
  unsigned o0x = wvu * (4u * KCX * 512u * 2u);
  unsigned o0h = wvu * (4u * KCH * 512u * 2u);
  unsigned o1x = wvu * (4u * KCH * 512u * 2u);
  unsigned o1h = wvu * (4u * KCH * 512u * 2u);

  for (int t = 0; t < Tn; ++t) {
    const int cur = t & 1, nxt = cur ^ 1;
    if (t + 1 < Tn) __builtin_prefetch(xrow + (size_t)(t + 1) * In, 0, 0);

    // Opaque re-definition each step: weight addresses become loop-variant so
    // the loads stay inside the loop (stream from L2) instead of being hoisted
    // and spilled to scratch. Values are unchanged (empty asm), SGPR-resident.
    asm volatile("" : "+s"(o0x), "+s"(o0h), "+s"(o1x), "+s"(o1h));
    const bf16_t* q0x = (const bf16_t*)((const char*)W0x + o0x);
    const bf16_t* q0h = (const bf16_t*)((const char*)W0h + o0h);
    const bf16_t* q1x = (const bf16_t*)((const char*)W1x + o1x);
    const bf16_t* q1h = (const bf16_t*)((const char*)W1h + o1h);

    // ================= layer 0 =================
    V16 Ax[KCX];
#pragma unroll
    for (int kc = 0; kc < KCX; ++kc) {
      const int K0 = kc * 32 + hi * 8;  // always < 80
      const int K1 = K0 + 16;
      Ax[kc].h[0] = cvt8(xrow + (size_t)t * In + K0);
      if (K1 < In) Ax[kc].h[1] = cvt8(xrow + (size_t)t * In + K1);
      else { v8bf z = {}; Ax[kc].h[1] = z; }  // K>=80 padding
    }
    V16 Ah[KCH];
    {
      const bf16_t* hb = &h0_lds[cur][nn * RS];
#pragma unroll
      for (int kc = 0; kc < KCH; ++kc) Ah[kc].v = ldA_lds(hb, kc, hi);
    }
    v8f acc[4];
#pragma unroll
    for (int g = 0; g < 4; ++g) {
      v8f a;
#pragma unroll
      for (int v = 0; v < 8; ++v) a[v] = bA0[g];
      acc[g] = a;
    }
    accum4<KCX>(acc, Ax, q0x, lane);
    accum4<KCH>(acc, Ah, q0h, lane);
    cell_update(acc, c0, &h0_lds[nxt][j], hi);
    __syncthreads();  // h0(t) visible to all waves

    // ================= layer 1 =================
    V16 Bx[KCH], Bh[KCH];
    {
      const bf16_t* hb0 = &h0_lds[nxt][nn * RS];
      const bf16_t* hb1 = &h1_lds[cur][nn * RS];
#pragma unroll
      for (int kc = 0; kc < KCH; ++kc) {
        Bx[kc].v = ldA_lds(hb0, kc, hi);
        Bh[kc].v = ldA_lds(hb1, kc, hi);
      }
    }
#pragma unroll
    for (int g = 0; g < 4; ++g) {
      v8f a;
#pragma unroll
      for (int v = 0; v < 8; ++v) a[v] = bA1[g];
      acc[g] = a;
    }
    accum4<KCH>(acc, Bx, q1x, lane);
    accum4<KCH>(acc, Bh, q1h, lane);
    cell_update(acc, c1, &h1_lds[nxt][j], hi);
    __syncthreads();  // h1(t) visible; also fences buffer reuse

    // ================= output head (O=1) =================
    if (wv == 0 && hi == 0) {
      const bf16_t* hr = &h1_lds[nxt][nn * RS];
      float s = bout0;
#pragma unroll
      for (int k8 = 0; k8 < Hn / 8; ++k8) {
        v8bf hv = *(const v8bf*)(hr + k8 * 8);
#pragma unroll
        for (int e = 0; e < 8; ++e) s += (float)hv[e] * wout[k8 * 8 + e];
      }
      out[(size_t)(b0 + nn) * Tn + t] = s > 0.0f ? s : 0.0f;
    }
  }
}

// ---------------- entry point ------------------------------------------------
extern "C" void kernel_launch(void* const* d_in, const int* in_sizes, int n_in,
                              void* d_out, int out_size, void* d_ws, size_t ws_size,
                              hipStream_t stream) {
  (void)in_sizes; (void)n_in; (void)out_size; (void)ws_size;
  const float* x     = (const float*)d_in[0];
  const float* W_ih0 = (const float*)d_in[1];
  const float* W_hh0 = (const float*)d_in[2];
  const float* b_ih0 = (const float*)d_in[3];
  const float* b_hh0 = (const float*)d_in[4];
  const float* W_ih1 = (const float*)d_in[5];
  const float* W_hh1 = (const float*)d_in[6];
  const float* b_ih1 = (const float*)d_in[7];
  const float* b_hh1 = (const float*)d_in[8];
  const float* W_out = (const float*)d_in[9];
  const float* b_out = (const float*)d_in[10];
  float* out = (float*)d_out;

  // workspace layout (bytes)
  char* ws = (char*)d_ws;
  bf16_t* W0xs = (bf16_t*)(ws + 0);        // 10*4*3*512*2  = 122880
  bf16_t* W0hs = (bf16_t*)(ws + 122880);   // 10*4*5*512*2  = 204800
  bf16_t* W1xs = (bf16_t*)(ws + 327680);
  bf16_t* W1hs = (bf16_t*)(ws + 532480);
  float*  bs0  = (float*)(ws + 737280);    // 640*4
  float*  bs1  = (float*)(ws + 739840);
  float*  wo   = (float*)(ws + 742400);    // 160*4
  float*  bo   = (float*)(ws + 743040);    // 4

  lstm_prep<<<208, 256, 0, stream>>>(W_ih0, W_hh0, b_ih0, b_hh0,
                                     W_ih1, W_hh1, b_ih1, b_hh1,
                                     W_out, b_out,
                                     W0xs, W0hs, W1xs, W1hs, bs0, bs1, wo, bo);

  lstm_persistent<<<Bn / 16, NT * 32, 0, stream>>>(x, W0xs, W0hs, W1xs, W1hs,
                                                   bs0, bs1, wo, bo, out);
}
